// DeepseekV2MLAAttentionFusion_62886911148243
// MI455X (gfx1250) — matrix operand order repaired and checked by reading, not compiled, not run on hardware
//
#include <hip/hip_runtime.h>
#include <hip/hip_bf16.h>

// ---------------------------------------------------------------------------
// DeepSeek-V2 MLA attention fusion for gfx1250 (MI455X).
// GEMM-like math through v_wmma_f32_16x16x32_bf16 (fp32 accumulate);
// attention K/V tiles staged into LDS via the Tensor Data Mover (TDM).
// ---------------------------------------------------------------------------

#define TSEQ  2048
#define HIDD  2048
#define NH    16
#define DNOPE 128
#define DROPE 64
#define DVAL  128
#define QLAT  1536
#define KVLAT 512
#define DQK   192      // DNOPE + DROPE
#define ATT_SCALE 0.07216878364870322f   // (192)^-0.5
#define LOG2_THETA 13.287712379549449f   // log2(10000)

typedef __bf16 bf16_t;
typedef __attribute__((ext_vector_type(16))) __bf16 bf16x16;
typedef __attribute__((ext_vector_type(8)))  __bf16 bf16x8;
typedef __attribute__((ext_vector_type(8)))  float  floatx8;
typedef unsigned int u32x4 __attribute__((ext_vector_type(4)));
typedef int          i32x4 __attribute__((ext_vector_type(4)));
typedef int          i32x8 __attribute__((ext_vector_type(8)));

#if defined(__has_builtin)
#if __has_builtin(__builtin_amdgcn_tensor_load_to_lds)
#define MLA_USE_TDM 1
#endif
#endif

// ---- WMMA fragment helpers (layouts per CDNA5 ISA 7.12.2, wave32) ----------

// A fragment: 16x32 bf16 tile from row-major matrix, p -> tile origin.
// lane l: row m = l%16; elems 0..7 = K base..base+7, elems 8..15 = K 16+base..
// with base = (l>=16) ? 8 : 0.
__device__ __forceinline__ bf16x16 frag_a_load(const bf16_t* p, int ld) {
    union { bf16x16 v; bf16x8 h[2]; } u;
    int l    = threadIdx.x & 31;
    int m    = l & 15;
    int base = (l & 16) ? 8 : 0;
    const bf16_t* q = p + (size_t)m * ld + base;
    u.h[0] = *(const bf16x8*)(q);
    u.h[1] = *(const bf16x8*)(q + 16);
    return u.v;
}

// B fragment: 32x16 bf16 (K x N). Source is W row-major [N,K] (i.e. B^T),
// p -> W tile origin [n0][k0]. lane l: col n = l%16; elems 0..7 = K koff..koff+7,
// elems 8..15 = K 16+koff.., with koff = (l>=16) ? 8 : 0.
__device__ __forceinline__ bf16x16 frag_b_load(const bf16_t* p, long long ld) {
    union { bf16x16 v; bf16x8 h[2]; } u;
    int l    = threadIdx.x & 31;
    int n    = l & 15;
    int koff = (l & 16) ? 8 : 0;
    const bf16_t* q = p + (long long)n * ld + koff;
    u.h[0] = *(const bf16x8*)(q);
    u.h[1] = *(const bf16x8*)(q + 16);
    return u.v;
}

__device__ __forceinline__ floatx8 wmma_bf16(bf16x16 a, bf16x16 b, floatx8 c) {
    return __builtin_amdgcn_wmma_f32_16x16x32_bf16(
        /*neg_a=*/false, a, /*neg_b=*/false, b,
        /*c_mod=*/(short)0, c, /*reuse_a=*/false, /*reuse_b=*/false);
}

// ---- TDM: 2D tile (global -> LDS), data_size = 2 bytes ---------------------
#ifdef MLA_USE_TDM
__device__ __forceinline__ void tdm_load_2d(const void* gsrc, void* ldst,
                                            unsigned tile0, unsigned tile1,
                                            unsigned stride0) {
    unsigned long long ga = (unsigned long long)(size_t)gsrc;
    unsigned lds = (unsigned)(size_t)ldst;           // LDS aperture: offset in [31:0]
    // D# group 0: count=1 | lds_addr | global_addr[56:0] | type=2
    u32x4 g0 = { 1u,
                 lds,
                 (unsigned)ga,
                 ((unsigned)(ga >> 32) & 0x01FFFFFFu) | (2u << 30) };
    // D# group 1: data_size=1(2B); tensor_dim0/1 = tile_dim0/1 (no clipping);
    // tensor_dim0_stride = stride0 elements.
    i32x8 g1 = { (int)(1u << 16),
                 (int)((tile0 & 0xFFFFu) << 16),                 // dim0[15:0] -> bits 63:48
                 (int)((tile0 >> 16) | ((tile1 & 0xFFFFu) << 16)),
                 (int)((tile1 >> 16) | (tile0 << 16)),           // tile_dim0 -> bits 127:112
                 (int)(tile1 & 0xFFFFu),                         // tile_dim1 (tile_dim2 = 0)
                 (int)stride0,
                 0, 0 };
    i32x4 z4 = { 0, 0, 0, 0 };
#if defined(__clang_major__) && __clang_major__ >= 23
    i32x8 z8 = { 0, 0, 0, 0, 0, 0, 0, 0 };
    __builtin_amdgcn_tensor_load_to_lds(g0, g1, z4, z4, z8, 0);
#else
    __builtin_amdgcn_tensor_load_to_lds(g0, g1, z4, z4, 0);
#endif
}
#endif

// ---- generic GEMM: C[M,N] (f32) = A[M,K] (bf16, row-major) @ W[N,K]^T ------
// block = 256 threads = 8 waves laid out 4(M) x 2(N); wave tile 32x32;
// block tile 128 x 64. grid = (N/64, M/128). K % 32 == 0.
__global__ __launch_bounds__(256)
void gemm_bf16_nt(const bf16_t* __restrict__ A, const bf16_t* __restrict__ W,
                  float* __restrict__ C, int M, int N, int K) {
    int wave = threadIdx.x >> 5;
    int wm   = wave >> 1;
    int wn   = wave & 1;
    int m0   = blockIdx.y * 128 + wm * 32;
    int n0   = blockIdx.x * 64  + wn * 32;

    floatx8 c00 = {}, c01 = {}, c10 = {}, c11 = {};
    const bf16_t* Ap0 = A + (size_t)m0 * K;
    const bf16_t* Ap1 = A + (size_t)(m0 + 16) * K;
    const bf16_t* Wp0 = W + (size_t)n0 * K;
    const bf16_t* Wp1 = W + (size_t)(n0 + 16) * K;

    for (int k0 = 0; k0 < K; k0 += 32) {
        bf16x16 a0 = frag_a_load(Ap0 + k0, K);
        bf16x16 a1 = frag_a_load(Ap1 + k0, K);
        bf16x16 b0 = frag_b_load(Wp0 + k0, K);
        bf16x16 b1 = frag_b_load(Wp1 + k0, K);
        c00 = wmma_bf16(a0, b0, c00);
        c01 = wmma_bf16(a0, b1, c01);
        c10 = wmma_bf16(a1, b0, c10);
        c11 = wmma_bf16(a1, b1, c11);
    }

    int lane  = threadIdx.x & 31;
    int half8 = (lane & 16) ? 8 : 0;
    int col   = lane & 15;
    float* Crow;
#pragma unroll
    for (int v = 0; v < 8; ++v) {
        Crow = C + (size_t)(m0 + half8 + v) * N;
        Crow[n0 + col]      = c00[v];
        Crow[n0 + 16 + col] = c01[v];
        Crow = C + (size_t)(m0 + 16 + half8 + v) * N;
        Crow[n0 + col]      = c10[v];
        Crow[n0 + 16 + col] = c11[v];
    }
}

// ---- elementwise / normalization helpers -----------------------------------

__global__ void f32_to_bf16_kernel(const float* __restrict__ x,
                                   bf16_t* __restrict__ y, long long n) {
    long long i = (long long)blockIdx.x * blockDim.x + threadIdx.x;
    if (i < n) y[i] = (bf16_t)x[i];
}

__global__ __launch_bounds__(256)
void rmsnorm_kernel(const float* __restrict__ X, int ld, int width,
                    const float* __restrict__ w, bf16_t* __restrict__ Y) {
    __shared__ float red[8];
    int t = blockIdx.x;
    const float* x = X + (size_t)t * ld;
    float ss = 0.f;
    for (int i = threadIdx.x; i < width; i += 256) { float v = x[i]; ss = fmaf(v, v, ss); }
#pragma unroll
    for (int o = 16; o > 0; o >>= 1) ss += __shfl_down(ss, o);
    if ((threadIdx.x & 31) == 0) red[threadIdx.x >> 5] = ss;
    __syncthreads();
    if (threadIdx.x == 0) {
        float s = 0.f;
#pragma unroll
        for (int i = 0; i < 8; ++i) s += red[i];
        red[0] = rsqrtf(s / (float)width + 1e-6f);
    }
    __syncthreads();
    float r = red[0];
    for (int i = threadIdx.x; i < width; i += 256)
        Y[(size_t)t * width + i] = (bf16_t)(x[i] * r * w[i]);
}

__device__ __forceinline__ float rope_invfreq(int j) {
    return exp2f(-(float)j * (LOG2_THETA / 32.0f));   // theta^(-2j/64)
}

__global__ void rope_kpe_kernel(const float* __restrict__ latent,
                                float* __restrict__ kpe) {
    int t = blockIdx.x;
    int j = threadIdx.x;           // 32 pairs
    float f = (float)t * rope_invfreq(j);
    float s, c; sincosf(f, &s, &c);
    const float* src = latent + (size_t)t * (KVLAT + DROPE) + KVLAT;
    float x1 = src[2 * j], x2 = src[2 * j + 1];
    kpe[(size_t)t * DROPE + 2 * j]     = x1 * c - x2 * s;
    kpe[(size_t)t * DROPE + 2 * j + 1] = x2 * c + x1 * s;
}

// q2[t][h*192+d] (f32) -> Qh[h][t][192] (bf16), rope on d in [128,192)
__global__ __launch_bounds__(256)
void pack_q_kernel(const float* __restrict__ q2, bf16_t* __restrict__ Qh) {
    int t = blockIdx.x;
    for (int i = threadIdx.x; i < NH * DQK; i += 256) {
        int h = i / DQK, d = i % DQK;
        const float* src = q2 + (size_t)t * (NH * DQK) + (size_t)h * DQK;
        float val;
        if (d < DNOPE) {
            val = src[d];
        } else {
            int j = (d - DNOPE) >> 1;
            float f = (float)t * rope_invfreq(j);
            float s, c; sincosf(f, &s, &c);
            float x1 = src[DNOPE + 2 * j], x2 = src[DNOPE + 2 * j + 1];
            val = ((d & 1) == 0) ? (x1 * c - x2 * s) : (x2 * c + x1 * s);
        }
        Qh[((size_t)h * TSEQ + t) * DQK + d] = (bf16_t)val;
    }
}

// kvr[t][h*256+d] (f32) + kpe -> Kh[h][t][192] (bf16), Vt[h][dv][t] (bf16)
__global__ __launch_bounds__(256)
void pack_kv_kernel(const float* __restrict__ kvr, const float* __restrict__ kpe,
                    bf16_t* __restrict__ Kh, bf16_t* __restrict__ Vt) {
    int t = blockIdx.x;
    for (int i = threadIdx.x; i < NH * 256; i += 256) {
        int h = i >> 8, d = i & 255;
        float val = kvr[(size_t)t * (NH * 256) + i];
        if (d < DNOPE)
            Kh[((size_t)h * TSEQ + t) * DQK + d] = (bf16_t)val;
        else
            Vt[((size_t)h * DVAL + (d - DNOPE)) * TSEQ + t] = (bf16_t)val;
    }
    for (int i = threadIdx.x; i < NH * DROPE; i += 256) {
        int h = i >> 6, j = i & 63;
        Kh[((size_t)h * TSEQ + t) * DQK + DNOPE + j] =
            (bf16_t)kpe[(size_t)t * DROPE + j];
    }
}

// ---- flash-style causal MLA attention --------------------------------------
// grid = (T/128, H); block = 256 (8 waves). Block owns 128 query rows of one
// head; K/V tiles of 32 keys are staged once per block into double-buffered
// LDS by the TDM (wave 0 issues tensor_load_to_lds, overlapped with compute),
// then all 8 waves consume them with WMMA.
__global__ __launch_bounds__(256)
void mla_attn_kernel(const bf16_t* __restrict__ Qh, const bf16_t* __restrict__ Kh,
                     const bf16_t* __restrict__ Vt, bf16_t* __restrict__ O) {
    __shared__ __align__(16) bf16_t Kbuf[2][32 * DQK];    // 2 x 12 KB
    __shared__ __align__(16) bf16_t Vbuf[2][DVAL * 32];   // 2 x 8 KB
    __shared__ __align__(16) bf16_t pbuf[8][16 * 32];     // 8 KB, per-wave P

    int wave = threadIdx.x >> 5;
    int lane = threadIdx.x & 31;
    int h    = blockIdx.y;
    int M0   = blockIdx.x * 128;
    int m0   = M0 + wave * 16;

    const bf16_t* Qbase = Qh + ((size_t)h * TSEQ + m0) * DQK;
    bf16x16 qa[6];
#pragma unroll
    for (int kk = 0; kk < 6; ++kk) qa[kk] = frag_a_load(Qbase + kk * 32, DQK);

    floatx8 acc[8];
#pragma unroll
    for (int i = 0; i < 8; ++i) acc[i] = (floatx8){};
    float mrow[8], lrow[8];
#pragma unroll
    for (int v = 0; v < 8; ++v) { mrow[v] = -1e30f; lrow[v] = 0.f; }

    int half8 = (lane & 16) ? 8 : 0;
    int col   = lane & 15;
    bf16_t* pb = pbuf[wave];

    const bf16_t* Kg = Kh + (size_t)h * TSEQ * DQK;   // + s0*DQK (rows contiguous)
    const bf16_t* Vg = Vt + (size_t)h * DVAL * TSEQ;  // + s0 (row stride TSEQ)

    // ---- prologue: stage tile s0 = 0 into buffer 0
#ifdef MLA_USE_TDM
    if (wave == 0) {
        tdm_load_2d(Kg, Kbuf[0], 32 * DQK, 1, 32 * DQK);   // contiguous 1-D tile
        tdm_load_2d(Vg, Vbuf[0], 32, DVAL, TSEQ);          // strided 2-D tile
        asm volatile("s_wait_tensorcnt 0x0" ::: "memory");
    }
#else
    for (int i = threadIdx.x; i < (32 * DQK) / 8; i += 256)
        ((bf16x8*)Kbuf[0])[i] = ((const bf16x8*)Kg)[i];
    for (int i = threadIdx.x; i < (DVAL * 32) / 8; i += 256)
        ((bf16x8*)Vbuf[0])[i] =
            *(const bf16x8*)(Vg + (size_t)(i >> 2) * TSEQ + (i & 3) * 8);
#endif
    __syncthreads();

    int nIter = (M0 + 128) / 32;
    int buf   = 0;
    for (int it = 0; it < nIter; ++it) {
        int s0 = it * 32;
        bool more = (it + 1 < nIter);

        // ---- issue next tile's DMA into the other buffer
        if (more) {
            const bf16_t* Kn = Kg + (size_t)(s0 + 32) * DQK;
            const bf16_t* Vn = Vg + (s0 + 32);
#ifdef MLA_USE_TDM
            if (wave == 0) {
                tdm_load_2d(Kn, Kbuf[buf ^ 1], 32 * DQK, 1, 32 * DQK);
                tdm_load_2d(Vn, Vbuf[buf ^ 1], 32, DVAL, TSEQ);
            }
#else
            for (int i = threadIdx.x; i < (32 * DQK) / 8; i += 256)
                ((bf16x8*)Kbuf[buf ^ 1])[i] = ((const bf16x8*)Kn)[i];
            for (int i = threadIdx.x; i < (DVAL * 32) / 8; i += 256)
                ((bf16x8*)Vbuf[buf ^ 1])[i] =
                    *(const bf16x8*)(Vn + (size_t)(i >> 2) * TSEQ + (i & 3) * 8);
#endif
        }

        if (s0 < m0 + 16) {   // causal: this wave still has live keys here
            // ---- scores: [16q x 32k] over d = 192
            floatx8 sc0 = {}, sc1 = {};
            const bf16_t* Kb = Kbuf[buf];
#pragma unroll
            for (int kk = 0; kk < 6; ++kk) {
                bf16x16 b0 = frag_b_load(Kb + kk * 32, DQK);
                bf16x16 b1 = frag_b_load(Kb + (size_t)16 * DQK + kk * 32, DQK);
                sc0 = wmma_bf16(qa[kk], b0, sc0);
                sc1 = wmma_bf16(qa[kk], b1, sc1);
            }

            // ---- scale + causal mask + online softmax
            float p0[8], p1[8], tmax[8];
#pragma unroll
            for (int v = 0; v < 8; ++v) {
                int row = m0 + half8 + v;
                float x0 = sc0[v] * ATT_SCALE;
                float x1 = sc1[v] * ATT_SCALE;
                if (s0 + col > row)      x0 = -1e30f;
                if (s0 + 16 + col > row) x1 = -1e30f;
                p0[v] = x0; p1[v] = x1;
                tmax[v] = fmaxf(x0, x1);
            }
#pragma unroll
            for (int v = 0; v < 8; ++v) {
                float tm = tmax[v];
                tm = fmaxf(tm, __shfl_xor(tm, 1));
                tm = fmaxf(tm, __shfl_xor(tm, 2));
                tm = fmaxf(tm, __shfl_xor(tm, 4));
                tm = fmaxf(tm, __shfl_xor(tm, 8));
                float nm   = fmaxf(mrow[v], tm);
                float corr = __expf(mrow[v] - nm);
                mrow[v] = nm;
                float e0 = __expf(p0[v] - nm);
                float e1 = __expf(p1[v] - nm);
                p0[v] = e0; p1[v] = e1;
                float rs = e0 + e1;
                rs += __shfl_xor(rs, 1);
                rs += __shfl_xor(rs, 2);
                rs += __shfl_xor(rs, 4);
                rs += __shfl_xor(rs, 8);
                lrow[v] = lrow[v] * corr + rs;
#pragma unroll
                for (int nt = 0; nt < 8; ++nt) acc[nt][v] *= corr;
            }

            // ---- stage P into LDS, re-load as 16x32 A fragment
#pragma unroll
            for (int v = 0; v < 8; ++v) {
                pb[(half8 + v) * 32 + col]      = (bf16_t)p0[v];
                pb[(half8 + v) * 32 + 16 + col] = (bf16_t)p1[v];
            }
            asm volatile("s_wait_dscnt 0" ::: "memory");   // same-wave LDS RAW
            bf16x16 pa = frag_a_load(pb, 32);

            // ---- out += P[16x32] @ V[32x128]
            const bf16_t* Vb = Vbuf[buf];
#pragma unroll
            for (int nt = 0; nt < 8; ++nt) {
                bf16x16 vb = frag_b_load(Vb + (size_t)nt * 16 * 32, 32);
                acc[nt] = wmma_bf16(pa, vb, acc[nt]);
            }
        }

#ifdef MLA_USE_TDM
        if (wave == 0 && more)
            asm volatile("s_wait_tensorcnt 0x0" ::: "memory");
#endif
        __syncthreads();
        buf ^= 1;
    }

    // ---- epilogue: normalize, write O[t][h*128+dv] as bf16
#pragma unroll
    for (int v = 0; v < 8; ++v) lrow[v] = 1.0f / lrow[v];
#pragma unroll
    for (int nt = 0; nt < 8; ++nt) {
#pragma unroll
        for (int v = 0; v < 8; ++v) {
            int row = m0 + half8 + v;
            int cdv = h * DVAL + nt * 16 + col;
            O[(size_t)row * (NH * DVAL) + cdv] = (bf16_t)(acc[nt][v] * lrow[v]);
        }
    }
}

// ---------------------------------------------------------------------------
extern "C" void kernel_launch(void* const* d_in, const int* in_sizes, int n_in,
                              void* d_out, int out_size, void* d_ws, size_t ws_size,
                              hipStream_t stream) {
    (void)in_sizes; (void)n_in; (void)out_size; (void)ws_size;
    const float* hs     = (const float*)d_in[1];
    const float* w_q_a  = (const float*)d_in[2];
    const float* q_ln   = (const float*)d_in[3];
    const float* w_q_b  = (const float*)d_in[4];
    const float* w_kv_a = (const float*)d_in[5];
    const float* kv_ln  = (const float*)d_in[6];
    const float* w_kv_b = (const float*)d_in[7];
    const float* w_o    = (const float*)d_in[8];
    float* out = (float*)d_out;

    char*  ws  = (char*)d_ws;
    size_t off = 0;
    auto alloc = [&](size_t bytes) -> void* {
        void* p = ws + off;
        off += (bytes + 255) & ~(size_t)255;
        return p;
    };

    bf16_t* hs_bf   = (bf16_t*)alloc((size_t)TSEQ * HIDD * 2);
    bf16_t* wqa_bf  = (bf16_t*)alloc((size_t)QLAT * HIDD * 2);
    bf16_t* wkva_bf = (bf16_t*)alloc((size_t)(KVLAT + DROPE) * HIDD * 2);
    bf16_t* wqb_bf  = (bf16_t*)alloc((size_t)(NH * DQK) * QLAT * 2);
    bf16_t* wkvb_bf = (bf16_t*)alloc((size_t)(NH * 256) * KVLAT * 2);
    bf16_t* wo_bf   = (bf16_t*)alloc((size_t)HIDD * (NH * DVAL) * 2);
    float*  rawbuf  = (float*)alloc((size_t)TSEQ * 4096 * 4);   // shared f32 GEMM out
    float*  latent  = (float*)alloc((size_t)TSEQ * (KVLAT + DROPE) * 4);
    bf16_t* qc_bf   = (bf16_t*)alloc((size_t)TSEQ * QLAT * 2);
    bf16_t* kvc_bf  = (bf16_t*)alloc((size_t)TSEQ * KVLAT * 2);
    float*  kpe     = (float*)alloc((size_t)TSEQ * DROPE * 4);
    bf16_t* Qhb     = (bf16_t*)alloc((size_t)NH * TSEQ * DQK * 2);
    bf16_t* Khb     = (bf16_t*)alloc((size_t)NH * TSEQ * DQK * 2);
    bf16_t* Vtb     = (bf16_t*)alloc((size_t)NH * DVAL * TSEQ * 2);
    bf16_t* attn_bf = (bf16_t*)alloc((size_t)TSEQ * NH * DVAL * 2);

    auto cvt = [&](const float* s, bf16_t* d, long long n) {
        f32_to_bf16_kernel<<<(unsigned)((n + 255) / 256), 256, 0, stream>>>(s, d, n);
    };
    cvt(hs,     hs_bf,   (long long)TSEQ * HIDD);
    cvt(w_q_a,  wqa_bf,  (long long)QLAT * HIDD);
    cvt(w_kv_a, wkva_bf, (long long)(KVLAT + DROPE) * HIDD);
    cvt(w_q_b,  wqb_bf,  (long long)(NH * DQK) * QLAT);
    cvt(w_kv_b, wkvb_bf, (long long)(NH * 256) * KVLAT);
    cvt(w_o,    wo_bf,   (long long)HIDD * (NH * DVAL));

    // 1) q_raw = hs @ w_q_a^T  [2048 x 1536]
    gemm_bf16_nt<<<dim3(QLAT / 64, TSEQ / 128), 256, 0, stream>>>(
        hs_bf, wqa_bf, rawbuf, TSEQ, QLAT, HIDD);
    rmsnorm_kernel<<<TSEQ, 256, 0, stream>>>(rawbuf, QLAT, QLAT, q_ln, qc_bf);

    // 2) latent = hs @ w_kv_a^T  [2048 x 576]
    gemm_bf16_nt<<<dim3((KVLAT + DROPE) / 64, TSEQ / 128), 256, 0, stream>>>(
        hs_bf, wkva_bf, latent, TSEQ, KVLAT + DROPE, HIDD);
    rmsnorm_kernel<<<TSEQ, 256, 0, stream>>>(latent, KVLAT + DROPE, KVLAT, kv_ln, kvc_bf);
    rope_kpe_kernel<<<TSEQ, 32, 0, stream>>>(latent, kpe);

    // 3) q = q_c @ w_q_b^T  [2048 x 3072] -> pack (rope q_pe) -> Qh
    gemm_bf16_nt<<<dim3((NH * DQK) / 64, TSEQ / 128), 256, 0, stream>>>(
        qc_bf, wqb_bf, rawbuf, TSEQ, NH * DQK, QLAT);
    pack_q_kernel<<<TSEQ, 256, 0, stream>>>(rawbuf, Qhb);

    // 4) kv = kv_c @ w_kv_b^T  [2048 x 4096] -> pack -> Kh, Vt
    gemm_bf16_nt<<<dim3((NH * 256) / 64, TSEQ / 128), 256, 0, stream>>>(
        kvc_bf, wkvb_bf, rawbuf, TSEQ, NH * 256, KVLAT);
    pack_kv_kernel<<<TSEQ, 256, 0, stream>>>(rawbuf, kpe, Khb, Vtb);

    // 5) causal attention -> attn_bf [2048 x 2048]
    mla_attn_kernel<<<dim3(TSEQ / 128, NH), 256, 0, stream>>>(Qhb, Khb, Vtb, attn_bf);

    // 6) out = attn @ w_o^T  [2048 x 2048] (f32 to d_out)
    gemm_bf16_nt<<<dim3(HIDD / 64, TSEQ / 128), 256, 0, stream>>>(
        attn_bf, wo_bf, out, TSEQ, HIDD, NH * DVAL);
}
